// GATv2WithGlobal_3143916060994
// MI455X (gfx1250) — matrix operand order repaired and checked by reading, not compiled, not run on hardware
//
#include <hip/hip_runtime.h>
#include <hip/hip_bf16.h>
#include <stdint.h>

typedef __attribute__((ext_vector_type(16))) __bf16 v16bf;
typedef __attribute__((ext_vector_type(8)))  __bf16 v8bf;
typedef __attribute__((ext_vector_type(2)))  __bf16 v2bf;
typedef __attribute__((ext_vector_type(8)))  float  v8f;

constexpr int Nn   = 50000;
constexpr int Ne   = 800000;
constexpr int Ng   = 256;
constexpr int Fdim = 9;
constexpr int GFdim= 187;
constexpr int Hh   = 4;
constexpr int O1c  = 64, O2c = 128;
constexpr int C1   = Hh * O1c;   // 256
constexpr int C2   = Hh * O2c;   // 512
constexpr int E2   = Ne + Nn;    // edges incl. self loops (850000, /8 exact)
constexpr float BN_EPS = 1e-5f;
constexpr float SM_EPS = 1e-16f;
constexpr int BK = 32;

__device__ __forceinline__ unsigned enc_f32(float f) {
  unsigned b = __float_as_uint(f);
  return (b & 0x80000000u) ? ~b : (b | 0x80000000u);  // order-preserving map
}
__device__ __forceinline__ float dec_f32(unsigned u) {
  unsigned b = (u & 0x80000000u) ? (u & 0x7FFFFFFFu) : ~u;
  return __uint_as_float(b);
}

// ---------------------------------------------------------------------------
// GEMM staging helpers.  GM/GK select row / K-tail guarding; the hot path
// (<false,false>) is branch-free with unconditional loads.
// ---------------------------------------------------------------------------
template<bool GM, bool GK, typename AT>
__device__ __forceinline__ void stage_A(__bf16 (&As)[128][40], const AT* __restrict__ A,
                                        int M, int K, int m0, int k0, int tid)
{
  constexpr bool ABF = (sizeof(AT) == 2);
#pragma unroll
  for (int i = 0; i < 8; ++i) {
    int idx = i * 256 + tid;            // 0..2047 (pairs)
    int r  = idx >> 4;                  // row 0..127
    int kk = (idx & 15) << 1;           // 0,2,..,30
    int gm = m0 + r, gk = k0 + kk;
    if constexpr (!GM && !GK) {
      const AT* ap = A + (size_t)gm * K + gk;
      if constexpr (ABF) {
        *(unsigned*)&As[r][kk] = *(const unsigned*)ap;   // raw bf16 pair copy
      } else {
        v2bf p; p[0] = (__bf16)ap[0]; p[1] = (__bf16)ap[1];
        *(v2bf*)&As[r][kk] = p;
      }
    } else {
      int gmc = (!GM || gm < M)     ? gm     : (M - 1);
      int g0  = (!GK || gk < K)     ? gk     : (K - 1);
      int g1  = (!GK || gk + 1 < K) ? gk + 1 : (K - 1);
      float f0 = (float)A[(size_t)gmc * K + g0];
      float f1 = (float)A[(size_t)gmc * K + g1];
      bool okm = (!GM || gm < M);
      f0 = (okm && (!GK || gk     < K)) ? f0 : 0.f;
      f1 = (okm && (!GK || gk + 1 < K)) ? f1 : 0.f;
      v2bf p; p[0] = (__bf16)f0; p[1] = (__bf16)f1;
      *(v2bf*)&As[r][kk] = p;
    }
  }
}

template<bool GK>
__device__ __forceinline__ void stage_B(__bf16 (&BsF)[4][32][24], const float* __restrict__ W,
                                        int K, int Nc, int n0, int k0, int tid)
{
#pragma unroll
  for (int i = 0; i < 4; ++i) {
    int idx = i * 256 + tid;            // 0..1023 (pairs)
    int cc = idx & 63;
    int kk = (idx >> 6) << 1;           // 0,2,..,30
    int gn = n0 + cc, gk = k0 + kk;     // n0+cc always < Nc (Nc multiple of 64)
    float f0, f1;
    if constexpr (!GK) {
      const float* wp = W + (size_t)gk * Nc + gn;
      f0 = wp[0]; f1 = wp[Nc];
    } else {
      int g0 = (gk < K)     ? gk     : (K - 1);
      int g1 = (gk + 1 < K) ? gk + 1 : (K - 1);
      f0 = W[(size_t)g0 * Nc + gn];
      f1 = W[(size_t)g1 * Nc + gn];
      f0 = (gk     < K) ? f0 : 0.f;
      f1 = (gk + 1 < K) ? f1 : 0.f;
    }
    int cb = cc >> 4;
    int ln = (cc & 15) + ((kk & 8) ? 16 : 0);
    int t  = (kk & 7) + ((kk & 16) ? 8 : 0);
    v2bf p; p[0] = (__bf16)f0; p[1] = (__bf16)f1;
    *(v2bf*)&BsF[cb][ln][t] = p;
  }
}

__device__ __forceinline__ void wmma_step(__bf16 (&As)[128][40], __bf16 (&BsF)[4][32][24],
                                          int wave, int lane, v8f (&acc)[4])
{
  // A fragment (ISA 16-bit A 16x32): two aligned ds_load_b128 + shuffle
  const int mr = wave * 16 + (lane & 15);
  const int kb = (lane & 16) ? 8 : 0;
  v8bf a0 = *(const v8bf*)&As[mr][kb];
  v8bf a1 = *(const v8bf*)&As[mr][kb + 16];
  v16bf af = __builtin_shufflevector(a0, a1, 0,1,2,3,4,5,6,7,8,9,10,11,12,13,14,15);
#pragma unroll
  for (int cb = 0; cb < 4; ++cb) {
    v8bf b0 = *(const v8bf*)&BsF[cb][lane][0];
    v8bf b1 = *(const v8bf*)&BsF[cb][lane][8];
    v16bf bfr = __builtin_shufflevector(b0, b1, 0,1,2,3,4,5,6,7,8,9,10,11,12,13,14,15);
    acc[cb] = __builtin_amdgcn_wmma_f32_16x16x32_bf16(
        false, af, false, bfr, (short)0, acc[cb], false, false);
  }
}

// ---------------------------------------------------------------------------
// Tiled WMMA GEMM: C[M,Nc] = A[M,K] * W[K,Nc] + bias, optional relu,
// output bf16 or f32.  Nc must be a multiple of 64 and fully covered by grid.
// Block = 256 threads = 8 waves; 128x64 tile; 4 WMMAs per wave per K-step.
// Hot path (interior M-block, full K-step) is completely guard-free.
// ---------------------------------------------------------------------------
template<typename AT, bool RELU, bool OUT_BF16>
__global__ __launch_bounds__(256)
void wmma_gemm_kernel(const AT* __restrict__ A, const float* __restrict__ W,
                      const float* __restrict__ bias, void* __restrict__ Out,
                      int M, int K, int Nc)
{
  __shared__ __align__(16) __bf16 As[128][40];       // 80B row stride
  __shared__ __align__(16) __bf16 BsF[4][32][24];    // fragment-major, 48B stride

  const int tid  = threadIdx.x;
  const int wave = tid >> 5;
  const int lane = tid & 31;
  const int m0 = blockIdx.y * 128;
  const int n0 = blockIdx.x * 64;

  v8f acc[4] = {};
  const int  kFull = K - (K % BK);
  const bool fullM = (m0 + 128 <= M);   // block-uniform

  if (fullM) {
    for (int k0 = 0; k0 < kFull; k0 += BK) {
      stage_A<false, false>(As, A, M, K, m0, k0, tid);
      stage_B<false>(BsF, W, K, Nc, n0, k0, tid);
      __syncthreads();
      wmma_step(As, BsF, wave, lane, acc);
      __syncthreads();
    }
    if (kFull < K) {
      stage_A<false, true>(As, A, M, K, m0, kFull, tid);
      stage_B<true>(BsF, W, K, Nc, n0, kFull, tid);
      __syncthreads();
      wmma_step(As, BsF, wave, lane, acc);
      __syncthreads();
    }
  } else {
    for (int k0 = 0; k0 < kFull; k0 += BK) {
      stage_A<true, false>(As, A, M, K, m0, k0, tid);
      stage_B<false>(BsF, W, K, Nc, n0, k0, tid);
      __syncthreads();
      wmma_step(As, BsF, wave, lane, acc);
      __syncthreads();
    }
    if (kFull < K) {
      stage_A<true, true>(As, A, M, K, m0, kFull, tid);
      stage_B<true>(BsF, W, K, Nc, n0, kFull, tid);
      __syncthreads();
      wmma_step(As, BsF, wave, lane, acc);
      __syncthreads();
    }
  }

  // C/D layout: lane%16 = N, VGPR j = M row (+8 for lanes 16-31)
  const int rbase = m0 + wave * 16 + ((lane & 16) ? 8 : 0);
  const int nlane = lane & 15;
#pragma unroll
  for (int cb = 0; cb < 4; ++cb) {
    int col = n0 + cb * 16 + nlane;            // always < Nc by launch contract
    float bv = bias ? bias[col] : 0.f;
#pragma unroll
    for (int j = 0; j < 8; ++j) {
      int row = rbase + j;
      if (!fullM && row >= M) continue;
      float val = acc[cb][j] + bv;
      if (RELU) val = fmaxf(val, 0.f);
      if (OUT_BF16) ((__bf16*)Out)[(size_t)row * Nc + col] = (__bf16)val;
      else          ((float*)Out)[(size_t)row * Nc + col] = val;
    }
  }
}

// ---------------------------------------------------------------------------
// Wave-per-edge attention logits: a[e,h] = sum_o att[h,o]*lrelu(xl[s]+xr[d])
// plus segment-max via order-preserving u32 atomicMax on dst.
// ---------------------------------------------------------------------------
template<int OSHIFT>
__global__ __launch_bounds__(256)
void edge_logits_kernel(const __bf16* __restrict__ xl, const __bf16* __restrict__ xr,
                        const int* __restrict__ src, const int* __restrict__ dst,
                        const float* __restrict__ att,
                        float* __restrict__ elog, unsigned* __restrict__ amaxEnc)
{
  constexpr int O = 1 << OSHIFT;
  constexpr int C = Hh * O;
  int wave = (blockIdx.x << 3) + (threadIdx.x >> 5);
  int lane = threadIdx.x & 31;
  if (wave >= E2) return;
  int s, d;
  if (wave < Ne) { s = src[wave]; d = dst[wave]; } else { s = d = wave - Ne; }
  const unsigned* rl = (const unsigned*)(xl + (size_t)s * C);  // packed bf16 pairs
  const unsigned* rr = (const unsigned*)(xr + (size_t)d * C);
  float acc[Hh] = {0.f, 0.f, 0.f, 0.f};
#pragma unroll
  for (int i = 0; i < C / 64; ++i) {
    int u = lane + 32 * i;          // h = (2u)>>OSHIFT is compile-time per i
    unsigned pl = rl[u], pr = rr[u];
    float l0 = __uint_as_float(pl << 16);
    float l1 = __uint_as_float(pl & 0xFFFF0000u);
    float r0 = __uint_as_float(pr << 16);
    float r1 = __uint_as_float(pr & 0xFFFF0000u);
    float e0 = l0 + r0; e0 = (e0 > 0.f) ? e0 : 0.2f * e0;
    float e1 = l1 + r1; e1 = (e1 > 0.f) ? e1 : 0.2f * e1;
    int elem = u << 1;
    int h = elem >> OSHIFT;
    int o = elem & (O - 1);
    acc[h] += e0 * att[h * O + o] + e1 * att[h * O + o + 1];
  }
#pragma unroll
  for (int h = 0; h < Hh; ++h)
#pragma unroll
    for (int off = 16; off > 0; off >>= 1)
      acc[h] += __shfl_xor(acc[h], off, 32);   // wave32 reduce
  if (lane == 0) {
#pragma unroll
    for (int h = 0; h < Hh; ++h) {
      elog[(size_t)wave * Hh + h] = acc[h];
      atomicMax(&amaxEnc[(size_t)d * Hh + h], enc_f32(acc[h]));
    }
  }
}

__global__ void edge_exp_kernel(const int* __restrict__ dst, float* __restrict__ elog,
                                const unsigned* __restrict__ amaxEnc, float* __restrict__ den)
{
  int idx = blockIdx.x * blockDim.x + threadIdx.x;
  if (idx >= E2 * Hh) return;
  int e = idx >> 2, h = idx & 3;
  int d = (e < Ne) ? dst[e] : (e - Ne);
  float amax = dec_f32(amaxEnc[(size_t)d * Hh + h]);
  float ex = expf(elog[idx] - amax);
  elog[idx] = ex;
  atomicAdd(&den[(size_t)d * Hh + h], ex);
}

template<int OSHIFT>
__global__ __launch_bounds__(256)
void edge_agg_kernel(const __bf16* __restrict__ xl, const int* __restrict__ src,
                     const int* __restrict__ dst, const float* __restrict__ elog,
                     const float* __restrict__ den, float* __restrict__ agg)
{
  constexpr int O = 1 << OSHIFT;
  constexpr int C = Hh * O;
  int wave = (blockIdx.x << 3) + (threadIdx.x >> 5);
  int lane = threadIdx.x & 31;
  if (wave >= E2) return;
  int s, d;
  if (wave < Ne) { s = src[wave]; d = dst[wave]; } else { s = d = wave - Ne; }
  float alpha[Hh];
#pragma unroll
  for (int h = 0; h < Hh; ++h)
    alpha[h] = elog[(size_t)wave * Hh + h] / (den[(size_t)d * Hh + h] + SM_EPS);
  const unsigned* rl = (const unsigned*)(xl + (size_t)s * C);
  float* ao = agg + (size_t)d * C;
#pragma unroll
  for (int i = 0; i < C / 64; ++i) {
    int u = lane + 32 * i;
    unsigned pl = rl[u];
    float l0 = __uint_as_float(pl << 16);
    float l1 = __uint_as_float(pl & 0xFFFF0000u);
    int elem = u << 1;
    int h = elem >> OSHIFT;
    atomicAdd(&ao[elem],     l0 * alpha[h]);   // global_atomic_add_f32, L2-resident
    atomicAdd(&ao[elem + 1], l1 * alpha[h]);
  }
}

__global__ void bias_bn_relu_kernel(const float* __restrict__ raw, const float* __restrict__ bias,
                                    const float* __restrict__ g, const float* __restrict__ b,
                                    const float* __restrict__ m, const float* __restrict__ v,
                                    __bf16* __restrict__ out, int C, int total)
{
  int idx = blockIdx.x * blockDim.x + threadIdx.x;
  if (idx >= total) return;
  int c = idx % C;
  float x = raw[idx] + bias[c];
  float y = g[c] * (x - m[c]) * rsqrtf(v[c] + BN_EPS) + b[c];
  out[idx] = (__bf16)fmaxf(y, 0.f);
}

__global__ __launch_bounds__(256)
void pool_sum_kernel(const __bf16* __restrict__ h2, const int* __restrict__ batch,
                     float* __restrict__ pooled)
{
  int node = blockIdx.x;
  int bg = batch[node];
  const __bf16* row = h2 + (size_t)node * C2;
  float* po = pooled + (size_t)bg * C2;
  for (int c = threadIdx.x; c < C2; c += 256)
    atomicAdd(&po[c], (float)row[c]);
}

__global__ void count_kernel(const int* __restrict__ batch, float* __restrict__ cnt)
{
  int i = blockIdx.x * blockDim.x + threadIdx.x;
  if (i < Nn) atomicAdd(&cnt[batch[i]], 1.f);
}

__global__ void build_zin_kernel(const float* __restrict__ pooled, const float* __restrict__ cnt,
                                 const float* __restrict__ gf, float* __restrict__ zin)
{
  constexpr int ZC = C2 + GFdim;  // 699
  int idx = blockIdx.x * blockDim.x + threadIdx.x;
  if (idx >= Ng * ZC) return;
  int g = idx / ZC, c = idx % ZC;
  float v;
  if (c < C2) v = pooled[(size_t)g * C2 + c] / fmaxf(cnt[g], 1.f);
  else        v = gf[(size_t)g * GFdim + (c - C2)];
  zin[idx] = v;
}

__global__ __launch_bounds__(256)
void fc2_kernel(const float* __restrict__ z, const float* __restrict__ w,
                const float* __restrict__ b, float* __restrict__ out)
{
  int g = (blockIdx.x << 3) + (threadIdx.x >> 5);
  int lane = threadIdx.x & 31;
  if (g >= Ng) return;
  const float* zr = z + (size_t)g * 128;
  float s = 0.f;
#pragma unroll
  for (int i = 0; i < 4; ++i) s += zr[lane + 32 * i] * w[lane + 32 * i];
#pragma unroll
  for (int off = 16; off > 0; off >>= 1) s += __shfl_xor(s, off, 32);
  if (lane == 0) out[g] = s + b[0];
}

// ---------------------------------------------------------------------------
extern "C" void kernel_launch(void* const* d_in, const int* in_sizes, int n_in,
                              void* d_out, int out_size, void* d_ws, size_t ws_size,
                              hipStream_t stream)
{
  (void)in_sizes; (void)n_in; (void)out_size; (void)ws_size;
  const float* x    = (const float*)d_in[0];
  const int*   ei   = (const int*)d_in[1];
  const int*   batch= (const int*)d_in[2];
  const float* gfeat= (const float*)d_in[3];
  const float* Wl1  = (const float*)d_in[4];  const float* bl1 = (const float*)d_in[5];
  const float* Wr1  = (const float*)d_in[6];  const float* br1 = (const float*)d_in[7];
  const float* att1 = (const float*)d_in[8];  const float* bias1=(const float*)d_in[9];
  const float* bn1g = (const float*)d_in[10]; const float* bn1b=(const float*)d_in[11];
  const float* bn1m = (const float*)d_in[12]; const float* bn1v=(const float*)d_in[13];
  const float* Wl2  = (const float*)d_in[14]; const float* bl2 = (const float*)d_in[15];
  const float* Wr2  = (const float*)d_in[16]; const float* br2 = (const float*)d_in[17];
  const float* att2 = (const float*)d_in[18]; const float* bias2=(const float*)d_in[19];
  const float* bn2g = (const float*)d_in[20]; const float* bn2b=(const float*)d_in[21];
  const float* bn2m = (const float*)d_in[22]; const float* bn2v=(const float*)d_in[23];
  const float* fc1w = (const float*)d_in[24]; const float* fc1b=(const float*)d_in[25];
  const float* fc2w = (const float*)d_in[26]; const float* fc2b=(const float*)d_in[27];
  const int* srcI = ei;
  const int* dstI = ei + Ne;

  char* p = (char*)d_ws;
  auto alloc = [&](size_t bytes) -> char* {
    char* r = p; p += (bytes + 255) & ~(size_t)255; return r;
  };
  float*    agg    = (float*)   alloc((size_t)Nn * C2 * 4);  // fp32 accumulator (reused)
  __bf16*   hbuf   = (__bf16*)  alloc((size_t)Nn * C2 * 2);  // h1 then h2 (bf16)
  __bf16*   xlb    = (__bf16*)  alloc((size_t)Nn * C2 * 2);  // xl (bf16, L2-resident)
  __bf16*   xrb    = (__bf16*)  alloc((size_t)Nn * C2 * 2);  // xr
  float*    elog   = (float*)   alloc((size_t)E2 * Hh * 4);  // logits -> exp
  unsigned* amax   = (unsigned*)alloc((size_t)Nn * Hh * 4);
  float*    den    = (float*)   alloc((size_t)Nn * Hh * 4);
  float*    pooled = (float*)   alloc((size_t)Ng * C2 * 4);
  float*    cnt    = (float*)   alloc((size_t)Ng * 4);
  float*    zin    = (float*)   alloc((size_t)Ng * (C2 + GFdim) * 4);
  float*    zmid   = (float*)   alloc((size_t)Ng * 128 * 4);

  dim3 blk(256);
  const int eblocks = E2 / 8;                 // wave-per-edge, 8 waves/block

  // ---- Layer 1: x @ {Wl1,Wr1} (+bias) -> bf16 -------------------------------
  {
    dim3 grid(C1 / 64, (Nn + 127) / 128);
    wmma_gemm_kernel<float, false, true><<<grid, blk, 0, stream>>>(x, Wl1, bl1, xlb, Nn, Fdim, C1);
    wmma_gemm_kernel<float, false, true><<<grid, blk, 0, stream>>>(x, Wr1, br1, xrb, Nn, Fdim, C1);
  }
  hipMemsetAsync(amax, 0, (size_t)Nn * Hh * 4, stream);
  hipMemsetAsync(den,  0, (size_t)Nn * Hh * 4, stream);
  hipMemsetAsync(agg,  0, (size_t)Nn * C1 * 4, stream);
  edge_logits_kernel<6><<<eblocks, blk, 0, stream>>>(xlb, xrb, srcI, dstI, att1, elog, amax);
  edge_exp_kernel<<<(E2 * Hh + 255) / 256, blk, 0, stream>>>(dstI, elog, amax, den);
  edge_agg_kernel<6><<<eblocks, blk, 0, stream>>>(xlb, srcI, dstI, elog, den, agg);
  bias_bn_relu_kernel<<<((Nn * C1) + 255) / 256, blk, 0, stream>>>(
      agg, bias1, bn1g, bn1b, bn1m, bn1v, hbuf, C1, Nn * C1);

  // ---- Layer 2: h1 @ {Wl2,Wr2} (bf16 WMMA) ---------------------------------
  {
    dim3 grid(C2 / 64, (Nn + 127) / 128);
    wmma_gemm_kernel<__bf16, false, true><<<grid, blk, 0, stream>>>(hbuf, Wl2, bl2, xlb, Nn, C1, C2);
    wmma_gemm_kernel<__bf16, false, true><<<grid, blk, 0, stream>>>(hbuf, Wr2, br2, xrb, Nn, C1, C2);
  }
  hipMemsetAsync(amax, 0, (size_t)Nn * Hh * 4, stream);
  hipMemsetAsync(den,  0, (size_t)Nn * Hh * 4, stream);
  hipMemsetAsync(agg,  0, (size_t)Nn * C2 * 4, stream);
  edge_logits_kernel<7><<<eblocks, blk, 0, stream>>>(xlb, xrb, srcI, dstI, att2, elog, amax);
  edge_exp_kernel<<<(E2 * Hh + 255) / 256, blk, 0, stream>>>(dstI, elog, amax, den);
  edge_agg_kernel<7><<<eblocks, blk, 0, stream>>>(xlb, srcI, dstI, elog, den, agg);
  bias_bn_relu_kernel<<<((Nn * C2) + 255) / 256, blk, 0, stream>>>(
      agg, bias2, bn2g, bn2b, bn2m, bn2v, hbuf, C2, Nn * C2);

  // ---- Pool + MLP head ------------------------------------------------------
  hipMemsetAsync(pooled, 0, (size_t)Ng * C2 * 4, stream);
  hipMemsetAsync(cnt,    0, (size_t)Ng * 4, stream);
  pool_sum_kernel<<<Nn, blk, 0, stream>>>(hbuf, batch, pooled);
  count_kernel<<<(Nn + 255) / 256, blk, 0, stream>>>(batch, cnt);
  build_zin_kernel<<<(Ng * (C2 + GFdim) + 255) / 256, blk, 0, stream>>>(pooled, cnt, gfeat, zin);
  {
    dim3 grid(128 / 64, (Ng + 127) / 128);
    wmma_gemm_kernel<float, true, false><<<grid, blk, 0, stream>>>(
        zin, fc1w, fc1b, zmid, Ng, C2 + GFdim, 128);
  }
  fc2_kernel<<<Ng / 8, blk, 0, stream>>>(zmid, fc2w, fc2b, (float*)d_out);
}
